// ColumnDatasetEncoder_37812892074328
// MI455X (gfx1250) — matrix operand order repaired and checked by reading, not compiled, not run on hardware
//
#include <hip/hip_runtime.h>

#define BN_EPS 1e-5f

typedef __attribute__((ext_vector_type(16))) _Float16 v16h;
typedef __attribute__((ext_vector_type(8)))  float    v8f;
typedef __attribute__((ext_vector_type(4)))  float    v4f;

// seg(i) = last j with ptr[j] <= i  (matches searchsorted(..., side="right") - 1)
__device__ __forceinline__ int find_seg(const int* __restrict__ ptr, int row, int B) {
  int lo = 0, hi = B;              // invariant: ptr[lo] <= row < ptr[hi]
  while (hi - lo > 1) {
    int mid = (lo + hi) >> 1;
    if (ptr[mid] <= row) lo = mid; else hi = mid;
  }
  return lo;
}

// ---------------------------------------------------------------------------
// Kernel 1: fold BN into weights/bias, pre-pack weights into WMMA B layout.
// B layout (16x16x32 f16): lanes 0-15: n = lane, K in {0..7,16..23} ; lanes 16-31:
// n = lane-16, K in {8..15,24..31} (matches A layout).
// ---------------------------------------------------------------------------
__global__ __launch_bounds__(256) void prep_kernel(
    const float* __restrict__ gamma1, const float* __restrict__ beta1,
    const float* __restrict__ rm1,    const float* __restrict__ rv1,
    const float* __restrict__ W1,     const float* __restrict__ b1,
    const float* __restrict__ gamma2, const float* __restrict__ beta2,
    const float* __restrict__ rm2,    const float* __restrict__ rv2,
    const float* __restrict__ W2,     const float* __restrict__ b2,
    _Float16* __restrict__ W1f, _Float16* __restrict__ W2f,
    float* __restrict__ b1f, float* __restrict__ b2f)
{
  int tid = threadIdx.x;
  // W1f: 4 n-tiles x 32 lanes x 16 halves  (D=32 -> single K step)
  if (tid < 128) {
    int t = tid >> 5, lane = tid & 31;
    int n  = t * 16 + (lane & 15);
    int kh = (lane >> 4) * 8;
    for (int j = 0; j < 16; ++j) {
      int k = kh + (j < 8 ? j : 8 + j);
      float s = gamma1[k] * rsqrtf(rv1[k] + BN_EPS);
      W1f[((size_t)t * 32 + lane) * 16 + j] = (_Float16)(s * W1[k * 64 + n]);
    }
  }
  // W2f: 8 groups (kstep*4 + n-tile) x 32 lanes x 16 halves  (H=64 -> 2 K steps)
  {
    int g = tid >> 5;                 // 0..7
    int ks = g >> 2, t = g & 3, lane = tid & 31;
    int n  = t * 16 + (lane & 15);
    int kh = ks * 32 + (lane >> 4) * 8;
    for (int j = 0; j < 16; ++j) {
      int k = kh + (j < 8 ? j : 8 + j);
      float s = gamma2[k] * rsqrtf(rv2[k] + BN_EPS);
      W2f[((size_t)g * 32 + lane) * 16 + j] = (_Float16)(s * W2[k * 64 + n]);
    }
  }
  if (tid < 64) {            // b1f
    int n = tid;
    float acc = b1[n];
    for (int k = 0; k < 32; ++k) {
      float s = gamma1[k] * rsqrtf(rv1[k] + BN_EPS);
      acc += (beta1[k] - rm1[k] * s) * W1[k * 64 + n];
    }
    b1f[n] = acc;
  } else if (tid < 128) {    // b2f
    int n = tid - 64;
    float acc = b2[n];
    for (int k = 0; k < 64; ++k) {
      float s = gamma2[k] * rsqrtf(rv2[k] + BN_EPS);
      acc += (beta2[k] - rm2[k] * s) * W2[k * 64 + n];
    }
    b2f[n] = acc;
  }
}

// ---------------------------------------------------------------------------
// Kernel 2 (hot, memory-bound): per wave, TWO 16-row tiles (32 rows) of x.
// Non-temporal x loads (read-once, 256MB stream). One WMMA per 16-col tile
// of H (K=32=D), B-tiles/bias shared across both row tiles. Segment sums of
// h and x via shuffle butterflies + f32 global atomics; when both tiles fall
// in one segment (~87%), reductions/atomics are merged (one batch / 32 rows).
// A layout (16x16x32 f16): lanes 0-15: M=lane, K in {0..7,16..23};
// lanes 16-31: M=lane-16, K in {8..15,24..31}.
// ---------------------------------------------------------------------------
__global__ __launch_bounds__(256) void row_mlp_seg_kernel(
    const float* __restrict__ x, const int* __restrict__ ptr,
    const _Float16* __restrict__ W1f, const float* __restrict__ b1f,
    float* __restrict__ accx, float* __restrict__ acch, int nRows, int B)
{
  int wave = blockIdx.x * (blockDim.x >> 5) + (threadIdx.x >> 5);
  int lane = threadIdx.x & 31;
  int r0 = wave * 32;
  if (r0 >= nRows) return;                 // wave-uniform (N is multiple of 32)
  int half = lane >> 4, m = lane & 15;
  int kbase = half * 8;

  int rowA = r0 + m;
  int rowB = r0 + 16 + m;
  int segA = find_seg(ptr, rowA, B);
  int segB = find_seg(ptr, rowB, B);

  // ---- non-temporal streaming loads of both row tiles ----
  const float* xrA = x + (size_t)rowA * 32 + kbase;
  const float* xrB = x + (size_t)rowB * 32 + kbase;
  float xa[16], xb[16];
  *(v4f*)(xa + 0)  = __builtin_nontemporal_load((const v4f*)(xrA + 0));
  *(v4f*)(xa + 4)  = __builtin_nontemporal_load((const v4f*)(xrA + 4));
  *(v4f*)(xa + 8)  = __builtin_nontemporal_load((const v4f*)(xrA + 16));
  *(v4f*)(xa + 12) = __builtin_nontemporal_load((const v4f*)(xrA + 20));
  *(v4f*)(xb + 0)  = __builtin_nontemporal_load((const v4f*)(xrB + 0));
  *(v4f*)(xb + 4)  = __builtin_nontemporal_load((const v4f*)(xrB + 4));
  *(v4f*)(xb + 8)  = __builtin_nontemporal_load((const v4f*)(xrB + 16));
  *(v4f*)(xb + 12) = __builtin_nontemporal_load((const v4f*)(xrB + 20));

  v16h a0, a1;
  #pragma unroll
  for (int j = 0; j < 16; ++j) { a0[j] = (_Float16)xa[j]; a1[j] = (_Float16)xb[j]; }

  // shared B-tiles and biases (L0-resident)
  v16h w[4];
  float bb[4];
  #pragma unroll
  for (int t = 0; t < 4; ++t) {
    w[t]  = *(const v16h*)(W1f + ((size_t)t * 32 + lane) * 16);
    bb[t] = b1f[t * 16 + m];
  }

  int sA0 = __shfl(segA, 0, 32), sA15 = __shfl(segA, 15, 32);
  int sB0 = __shfl(segB, 0, 32), sB15 = __shfl(segB, 15, 32);
  bool uniA = (sA0 == sA15), uniB = (sB0 == sB15);
  bool merged = uniA && uniB && (sA0 == sB0);

  v8f cz = {};
  float hsA[4], hsB[4];

  // ---- tile A: WMMA + bias/ReLU + row-reduce (or per-row atomics) ----
  {
    v8f c[4];
    #pragma unroll
    for (int t = 0; t < 4; ++t)
      c[t] = __builtin_amdgcn_wmma_f32_16x16x32_f16(
          false, a0, false, w[t], (short)0, cz, false, false);
    if (uniA) {
      #pragma unroll
      for (int t = 0; t < 4; ++t) {
        float s = 0.0f;
        #pragma unroll
        for (int i = 0; i < 8; ++i) s += fmaxf(c[t][i] + bb[t], 0.0f);
        s += __shfl_xor(s, 16, 32);        // combine row groups 0-7 / 8-15
        hsA[t] = s;
      }
    } else {                               // rare: boundary inside tile
      #pragma unroll
      for (int i = 0; i < 8; ++i) {
        int sr = __shfl(segA, i + half * 8, 32);
        #pragma unroll
        for (int t = 0; t < 4; ++t)
          atomicAdd(&acch[(size_t)sr * 64 + t * 16 + m],
                    fmaxf(c[t][i] + bb[t], 0.0f));
      }
    }
  }
  // ---- tile B ----
  {
    v8f c[4];
    #pragma unroll
    for (int t = 0; t < 4; ++t)
      c[t] = __builtin_amdgcn_wmma_f32_16x16x32_f16(
          false, a1, false, w[t], (short)0, cz, false, false);
    if (uniB) {
      #pragma unroll
      for (int t = 0; t < 4; ++t) {
        float s = 0.0f;
        #pragma unroll
        for (int i = 0; i < 8; ++i) s += fmaxf(c[t][i] + bb[t], 0.0f);
        s += __shfl_xor(s, 16, 32);
        hsB[t] = s;
      }
    } else {
      #pragma unroll
      for (int i = 0; i < 8; ++i) {
        int sr = __shfl(segB, i + half * 8, 32);
        #pragma unroll
        for (int t = 0; t < 4; ++t)
          atomicAdd(&acch[(size_t)sr * 64 + t * 16 + m],
                    fmaxf(c[t][i] + bb[t], 0.0f));
      }
    }
  }

  // ---- emit h segment sums ----
  if (merged) {
    if (lane < 16) {
      #pragma unroll
      for (int t = 0; t < 4; ++t)
        atomicAdd(&acch[(size_t)sA0 * 64 + t * 16 + lane], hsA[t] + hsB[t]);
    }
  } else {
    if (uniA && lane < 16) {
      #pragma unroll
      for (int t = 0; t < 4; ++t)
        atomicAdd(&acch[(size_t)sA0 * 64 + t * 16 + lane], hsA[t]);
    }
    if (uniB && lane < 16) {
      #pragma unroll
      for (int t = 0; t < 4; ++t)
        atomicAdd(&acch[(size_t)sB0 * 64 + t * 16 + lane], hsB[t]);
    }
  }

  // ---- x segment sums (mean_x branch), f32 exact ----
  if (merged) {
    #pragma unroll
    for (int j = 0; j < 16; ++j) xa[j] += xb[j];
    #pragma unroll
    for (int mask = 1; mask <= 8; mask <<= 1) {
      #pragma unroll
      for (int j = 0; j < 16; ++j) xa[j] += __shfl_xor(xa[j], mask, 32);
    }
    if (m == 0) {
      #pragma unroll
      for (int j = 0; j < 16; ++j) {
        int k = kbase + (j < 8 ? j : 8 + j);   // j>=8 -> kbase+16+(j-8)
        atomicAdd(&accx[(size_t)sA0 * 32 + k], xa[j]);
      }
    }
  } else {
    if (uniA) {
      #pragma unroll
      for (int mask = 1; mask <= 8; mask <<= 1) {
        #pragma unroll
        for (int j = 0; j < 16; ++j) xa[j] += __shfl_xor(xa[j], mask, 32);
      }
      if (m == 0) {
        #pragma unroll
        for (int j = 0; j < 16; ++j) {
          int k = kbase + (j < 8 ? j : 8 + j);
          atomicAdd(&accx[(size_t)sA0 * 32 + k], xa[j]);
        }
      }
    } else {
      #pragma unroll
      for (int j = 0; j < 16; ++j) {
        int k = kbase + (j < 8 ? j : 8 + j);
        atomicAdd(&accx[(size_t)segA * 32 + k], xa[j]);
      }
    }
    if (uniB) {
      #pragma unroll
      for (int mask = 1; mask <= 8; mask <<= 1) {
        #pragma unroll
        for (int j = 0; j < 16; ++j) xb[j] += __shfl_xor(xb[j], mask, 32);
      }
      if (m == 0) {
        #pragma unroll
        for (int j = 0; j < 16; ++j) {
          int k = kbase + (j < 8 ? j : 8 + j);
          atomicAdd(&accx[(size_t)sB0 * 32 + k], xb[j]);
        }
      }
    } else {
      #pragma unroll
      for (int j = 0; j < 16; ++j) {
        int k = kbase + (j < 8 ? j : 8 + j);
        atomicAdd(&accx[(size_t)segB * 32 + k], xb[j]);
      }
    }
  }
}

// ---------------------------------------------------------------------------
// Kernel 3: per wave, 16 segments. mean_h -> f16, 2 K-step WMMA vs W2f,
// bias + ReLU, scatter to out[:, 32:96].
// ---------------------------------------------------------------------------
__global__ __launch_bounds__(256) void seg_mlp2_kernel(
    const float* __restrict__ acch, const int* __restrict__ ptr,
    const _Float16* __restrict__ W2f, const float* __restrict__ b2f,
    float* __restrict__ out, int B)
{
  int wave = blockIdx.x * (blockDim.x >> 5) + (threadIdx.x >> 5);
  int lane = threadIdx.x & 31;
  int s0 = wave * 16;
  if (s0 >= B) return;
  int half = lane >> 4, m = lane & 15;
  int srow = s0 + m;
  int cnt  = ptr[srow + 1] - ptr[srow];
  float inv = 1.0f / fmaxf((float)cnt, 1.0f);
  const float* hr = acch + (size_t)srow * 64;

  v16h a[2];
  #pragma unroll
  for (int ks = 0; ks < 2; ++ks) {
    int kb = ks * 32 + half * 8;
    float mv[16];
    *(float4*)(mv + 0)  = *(const float4*)(hr + kb + 0);
    *(float4*)(mv + 4)  = *(const float4*)(hr + kb + 4);
    *(float4*)(mv + 8)  = *(const float4*)(hr + kb + 16);
    *(float4*)(mv + 12) = *(const float4*)(hr + kb + 20);
    #pragma unroll
    for (int j = 0; j < 16; ++j) a[ks][j] = (_Float16)(mv[j] * inv);
  }

  #pragma unroll
  for (int t = 0; t < 4; ++t) {
    v8f c = {};
    v16h b0 = *(const v16h*)(W2f + ((size_t)(0 * 4 + t) * 32 + lane) * 16);
    c = __builtin_amdgcn_wmma_f32_16x16x32_f16(
        false, a[0], false, b0, (short)0, c, false, false);
    v16h b1v = *(const v16h*)(W2f + ((size_t)(1 * 4 + t) * 32 + lane) * 16);
    c = __builtin_amdgcn_wmma_f32_16x16x32_f16(
        false, a[1], false, b1v, (short)0, c, false, false);
    float bbv = b2f[t * 16 + m];
    #pragma unroll
    for (int i = 0; i < 8; ++i) {
      int rr = s0 + i + half * 8;
      out[(size_t)rr * 96 + 32 + t * 16 + m] = fmaxf(c[i] + bbv, 0.0f);
    }
  }
}

// ---------------------------------------------------------------------------
// Kernel 4: out[:, 0:32] = acc_x / denom
// ---------------------------------------------------------------------------
__global__ __launch_bounds__(256) void meanx_kernel(
    const float* __restrict__ accx, const int* __restrict__ ptr,
    float* __restrict__ out, int B)
{
  int idx = blockIdx.x * blockDim.x + threadIdx.x;
  if (idx >= B * 32) return;
  int b = idx >> 5, d = idx & 31;
  int cnt = ptr[b + 1] - ptr[b];
  out[(size_t)b * 96 + d] = accx[idx] / fmaxf((float)cnt, 1.0f);
}

extern "C" void kernel_launch(void* const* d_in, const int* in_sizes, int n_in,
                              void* d_out, int out_size, void* d_ws, size_t ws_size,
                              hipStream_t stream) {
  const float* x      = (const float*)d_in[0];
  const int*   ptr    = (const int*)  d_in[1];
  const float* gamma1 = (const float*)d_in[2];
  const float* beta1  = (const float*)d_in[3];
  const float* rm1    = (const float*)d_in[4];
  const float* rv1    = (const float*)d_in[5];
  const float* W1     = (const float*)d_in[6];
  const float* b1     = (const float*)d_in[7];
  const float* gamma2 = (const float*)d_in[8];
  const float* beta2  = (const float*)d_in[9];
  const float* rm2    = (const float*)d_in[10];
  const float* rv2    = (const float*)d_in[11];
  const float* W2     = (const float*)d_in[12];
  const float* b2     = (const float*)d_in[13];

  const int N = in_sizes[0] / 32;       // 2,097,152
  const int B = in_sizes[1] - 1;        // 8,192

  // Workspace layout (all offsets 32B-aligned):
  char* ws = (char*)d_ws;
  float*    accx = (float*)ws;                                   // B*32 f32
  float*    acch = (float*)(ws + (size_t)B * 32 * 4);            // B*64 f32
  _Float16* W1f  = (_Float16*)(ws + (size_t)B * 96 * 4);         // 2048 f16
  _Float16* W2f  = (_Float16*)((char*)W1f + 4096);               // 4096 f16
  float*    b1f  = (float*)((char*)W2f + 8192);                  // 64 f32
  float*    b2f  = b1f + 64;                                     // 64 f32

  // zero accumulators (graph-capturable)
  hipMemsetAsync(d_ws, 0, (size_t)B * 96 * 4, stream);

  prep_kernel<<<1, 256, 0, stream>>>(gamma1, beta1, rm1, rv1, W1, b1,
                                     gamma2, beta2, rm2, rv2, W2, b2,
                                     W1f, W2f, b1f, b2f);

  int waves2  = (N + 31) / 32;          // 65,536 wave-tiles (32 rows each)
  int blocks2 = (waves2 + 7) / 8;       // 8 waves / 256-thread block
  row_mlp_seg_kernel<<<blocks2, 256, 0, stream>>>(x, ptr, W1f, b1f,
                                                  accx, acch, N, B);

  int waves3  = (B + 15) / 16;          // 512
  int blocks3 = (waves3 + 7) / 8;       // 64
  seg_mlp2_kernel<<<blocks3, 256, 0, stream>>>(acch, ptr, W2f, b2f,
                                               (float*)d_out, B);

  meanx_kernel<<<(B * 32 + 255) / 256, 256, 0, stream>>>(accx, ptr,
                                                         (float*)d_out, B);
}
